// PWCNet_18923625906843
// MI455X (gfx1250) — compile-verified, matrix operand order checked by reading, not run on hardware
//
#include <hip/hip_runtime.h>
#include <hip/hip_bf16.h>
#include <stdint.h>

// ---------------------------------------------------------------------------
// PWC-Net forward for MI455X (gfx1250).
// Convs run as implicit-GEMM on V_WMMA_F32_16X16X32_F16 (f16 in, f32 acc).
// Per block: 32(Cout) x 128(pixels); per wave: 16x32 via 2 WMMAs sharing the
// A fragment. Double-buffered LDS pipeline, fragment-native LDS layout
// (ds_load_b128), pair-packed ds_store_b32 staging.
// Only the forward-flow branch is computed: the returned output
// upsample(flow_f)*20 provably never depends on the backward branch.
// ---------------------------------------------------------------------------

typedef __attribute__((ext_vector_type(16))) _Float16 v16h;
typedef __attribute__((ext_vector_type(8)))  _Float16 v8h;
typedef __attribute__((ext_vector_type(2)))  _Float16 v2h;
typedef __attribute__((ext_vector_type(8)))  float    v8f;

#define LRELU(v) ((v) >= 0.f ? (v) : 0.1f * (v))

// K-slot permutation: store K order [0..7,16..23,8..15,24..31] so a lane's
// 16 fragment halves are contiguous (swap bits 3 and 4 of k).
// Even k -> even slot, and kslot(k+1) == kslot(k)+1 (bit0 passes through).
__device__ __forceinline__ int kslot(int k)
{
    return (k & 7) | ((k & 8) << 1) | ((k & 16) >> 1);
}

// ============================ WMMA conv3x3 =================================
// GEMM view: M = Cout, N = B*Hout*Wout, K = Cin*9.
// Block: 256 threads = 8 waves as 2(M) x 4(N); wave tile = 16 x 32 (2 WMMAs).
#define AST 40   // LDS row stride in halves (80B, 16B-aligned)
__global__ __launch_bounds__(256)
void conv3x3_wmma(const float* __restrict__ x, const float* __restrict__ w,
                  const float* __restrict__ bias, float* __restrict__ out,
                  int B, int Cin, int H, int W, int Cout, int CoutTot, int chOff,
                  int Hout, int Wout, int stride, int dil, int dolrelu)
{
    __shared__ __align__(16) _Float16 sA[2][32 * AST];   // [co][kslot(ci)]
    __shared__ __align__(16) _Float16 sB[2][128 * AST];  // [pix][kslot(ci)]

    const int tid = threadIdx.x;
    const int lane = tid & 31;
    const int wave = tid >> 5;
    const int wm = wave >> 2;            // M sub-tile 0..1
    const int wn = wave & 3;             // N sub-tile 0..3 (32 pixels each)
    const int half = lane >> 4;          // K-split lane group per ISA layout
    const int l16 = lane & 15;

    const int coBase = blockIdx.y * 32;
    const int pixBase = blockIdx.x * 128;
    const int HW = Hout * Wout;
    const int Npix = B * HW;
    const int pad = dil;                 // k=3 => pad = dil

    // ---- per-thread staging geometry (loop-invariant, hoisted) ------------
    // A: thread covers co = coBase + (tid>>3); ci pairs 2*((tid&7)+8r), r=0,1
    const int coA = tid >> 3;            // 0..31
    const int prA = tid & 7;             // pair base
    // B: thread covers pixel p = tid&127; ci pairs 2*(cg+2r), cg=tid>>7, r=0..7
    const int pB = tid & 127;
    const int cg = tid >> 7;             // 0..1

    const int nB = pixBase + pB;
    const bool pvalid = nB < Npix;
    const int nBc = pvalid ? nB : 0;
    const int bi = nBc / HW;
    const int rem = nBc % HW;
    const int oy = rem / Wout, ox = rem % Wout;
    const size_t xImg = (size_t)bi * Cin * H * W;

    const int nCh = (Cin + 31) >> 5;
    const int total = 9 * nCh;

    // unconditional clamped loads + select-to-zero: lets the compiler batch
    // all 20 global loads per step with a single wait.
    auto loadA = [&](int tap, int cb, float (&av)[4]) {
#pragma unroll
        for (int r = 0; r < 2; ++r)
#pragma unroll
            for (int e = 0; e < 2; ++e) {
                int co = coBase + coA;
                int ci = cb + 2 * (prA + 8 * r) + e;
                bool v = (co < Cout) && (ci < Cin);
                int coc = co < Cout ? co : Cout - 1;
                int cic = ci < Cin ? ci : Cin - 1;
                float t = w[((size_t)coc * Cin + cic) * 9 + tap];
                av[2 * r + e] = v ? t : 0.f;
            }
    };
    auto loadB = [&](int tap, int cb, float (&bv)[16]) {
        int ky = tap / 3, kx = tap % 3;
        int iy = oy * stride - pad + ky * dil;
        int ix = ox * stride - pad + kx * dil;
        bool inb = pvalid && iy >= 0 && iy < H && ix >= 0 && ix < W;
        int iyc = min(max(iy, 0), H - 1);
        int ixc = min(max(ix, 0), W - 1);
        size_t base = xImg + (size_t)iyc * W + ixc;
#pragma unroll
        for (int r = 0; r < 8; ++r)
#pragma unroll
            for (int e = 0; e < 2; ++e) {
                int ci = cb + 2 * (cg + 2 * r) + e;
                bool v = inb && (ci < Cin);
                int cic = ci < Cin ? ci : Cin - 1;
                float t = x[base + (size_t)cic * H * W];
                bv[2 * r + e] = v ? t : 0.f;
            }
    };
    // pair-packed b32 stores into the fragment-native layout
    auto stage = [&](int buf, const float (&av)[4], const float (&bv)[16]) {
#pragma unroll
        for (int r = 0; r < 2; ++r) {
            v2h p2; p2[0] = (_Float16)av[2 * r]; p2[1] = (_Float16)av[2 * r + 1];
            *(v2h*)&sA[buf][coA * AST + kslot(2 * (prA + 8 * r))] = p2;
        }
#pragma unroll
        for (int r = 0; r < 8; ++r) {
            v2h p2; p2[0] = (_Float16)bv[2 * r]; p2[1] = (_Float16)bv[2 * r + 1];
            *(v2h*)&sB[buf][pB * AST + kslot(2 * (cg + 2 * r))] = p2;
        }
    };

    v8f acc0 = {}, acc1 = {};
    float av[4], bv[16];
    loadA(0, 0, av);
    loadB(0, 0, bv);
    stage(0, av, bv);

    int tap = 0, cb = 0;
    for (int it = 0; it < total; ++it) {
        int ntap = tap, ncb = cb + 32;
        if (ncb >= Cin) { ncb = 0; ntap = tap + 1; }
        const bool more = (it + 1 < total);
        float avN[4], bvN[16];
        if (more) { loadA(ntap, ncb, avN); loadB(ntap, ncb, bvN); } // in flight over WMMAs

        __syncthreads();                                   // buffer it&1 ready

        // fragment loads: contiguous aligned ds_load_b128 pairs
        const int bufc = it & 1;
        const _Float16* pa  = &sA[bufc][(wm * 16 + l16) * AST + half * 16];
        const _Float16* pb0 = &sB[bufc][(wn * 32 + l16) * AST + half * 16];
        const _Float16* pb1 = pb0 + 16 * AST;
        v8h alo = *(const v8h*)pa,        ahi = *(const v8h*)(pa + 8);
        v8h b0l = *(const v8h*)pb0,       b0h = *(const v8h*)(pb0 + 8);
        v8h b1l = *(const v8h*)pb1,       b1h = *(const v8h*)(pb1 + 8);
        v16h a  = __builtin_shufflevector(alo, ahi, 0, 1, 2, 3, 4, 5, 6, 7,
                                          8, 9, 10, 11, 12, 13, 14, 15);
        v16h b0 = __builtin_shufflevector(b0l, b0h, 0, 1, 2, 3, 4, 5, 6, 7,
                                          8, 9, 10, 11, 12, 13, 14, 15);
        v16h b1 = __builtin_shufflevector(b1l, b1h, 0, 1, 2, 3, 4, 5, 6, 7,
                                          8, 9, 10, 11, 12, 13, 14, 15);
        acc0 = __builtin_amdgcn_wmma_f32_16x16x32_f16(
            false, a, false, b0, (short)0, acc0, false, false);
        acc1 = __builtin_amdgcn_wmma_f32_16x16x32_f16(
            false, a, false, b1, (short)0, acc1, false, false);

        if (more) stage((it + 1) & 1, avN, bvN);           // fill other buffer
        tap = ntap; cb = ncb;
    }

    // ---- store D: VGPR r -> M = r + 8*half ; col = lane%16 ----------------
#pragma unroll
    for (int sub = 0; sub < 2; ++sub) {
        const v8f& acc = sub ? acc1 : acc0;
#pragma unroll
        for (int r = 0; r < 8; ++r) {
            int m = r + 8 * half;
            int co = coBase + wm * 16 + m;
            int n = pixBase + wn * 32 + sub * 16 + l16;
            if (co < Cout && n < Npix) {
                int bi2 = n / HW;
                int rem2 = n % HW;
                int oy2 = rem2 / Wout, ox2 = rem2 % Wout;
                float v = acc[r] + bias[co];
                if (dolrelu) v = LRELU(v);
                out[(((size_t)bi2 * CoutTot + chOff + co) * Hout + oy2) * Wout + ox2] = v;
            }
        }
    }
}

// ============================ elementwise ==================================
__global__ void fill_kernel(float* p, float v, int n)
{
    int i = blockIdx.x * 256 + threadIdx.x;
    if (i < n) p[i] = v;
}

__global__ void add_kernel(float* a, const float* __restrict__ b, int n)
{
    int i = blockIdx.x * 256 + threadIdx.x;
    if (i < n) a[i] += b[i];
}

// copy src (B,C,H,W) into dst channels [off, off+C) of (B,Cd,H,W)
__global__ void chancopy_kernel(const float* __restrict__ src, float* __restrict__ dst,
                                int C, int H, int W, int Cd, int off, int n)
{
    int i = blockIdx.x * 256 + threadIdx.x;
    if (i >= n) return;
    int xw = i % W; int t = i / W;
    int yh = t % H; t /= H;
    int c = t % C;  int b = t / C;
    dst[(((size_t)b * Cd + off + c) * H + yh) * W + xw] = src[i];
}

// align-corners bilinear resize (matches jnp.linspace mapping), fused scale
__global__ void upsample_kernel(const float* __restrict__ src, float* __restrict__ dst,
                                int C, int H, int W, int H2, int W2, float scale, int n)
{
    int i = blockIdx.x * 256 + threadIdx.x;
    if (i >= n) return;
    int x2 = i % W2; int t = i / W2;
    int y2 = t % H2; t /= H2;
    int c = t % C;   int b = t / C;
    float ys = (H2 > 1) ? y2 * (float)(H - 1) / (float)(H2 - 1) : 0.f;
    float xs = (W2 > 1) ? x2 * (float)(W - 1) / (float)(W2 - 1) : 0.f;
    int y0 = (int)floorf(ys); int y1 = min(y0 + 1, H - 1); float wy = ys - y0;
    int x0 = (int)floorf(xs); int x1 = min(x0 + 1, W - 1); float wx = xs - x0;
    const float* pl = src + ((size_t)b * C + c) * H * W;
    float v = (pl[y0 * W + x0] * (1.f - wy) + pl[y1 * W + x0] * wy) * (1.f - wx)
            + (pl[y0 * W + x1] * (1.f - wy) + pl[y1 * W + x1] * wy) * wx;
    dst[i] = v * scale;
}

// grid-sample warp with the reference's corner-validity mask.
__global__ void warp_kernel(const float* __restrict__ x, const float* __restrict__ flow,
                            float* __restrict__ out, int C, int H, int W,
                            float coefx, float coefy, int n)
{
    int i = blockIdx.x * 256 + threadIdx.x;
    if (i >= n) return;
    int xx = i % W; int t = i / W;
    int yy = t % H; t /= H;
    int c = t % C;  int b = t / C;
    float fx = flow[(((size_t)b * 2 + 0) * H + yy) * W + xx];
    float fy = flow[(((size_t)b * 2 + 1) * H + yy) * W + xx];
    float px = xx + fx * coefx;
    float py = yy + fy * coefy;
    float x0 = floorf(px), y0 = floorf(py);
    float wx = px - x0, wy = py - y0;
    const float* pl = x + ((size_t)b * C + c) * H * W;
    float acc = 0.f, mask = 0.f;
#pragma unroll
    for (int dy = 0; dy < 2; ++dy)
#pragma unroll
        for (int dx = 0; dx < 2; ++dx) {
            float xi = x0 + dx, yi = y0 + dy;
            float wgt = (dx ? wx : 1.f - wx) * (dy ? wy : 1.f - wy);
            float valid = (xi >= 0.f && xi <= (float)(W - 1) &&
                           yi >= 0.f && yi <= (float)(H - 1)) ? 1.f : 0.f;
            float xc = fminf(fmaxf(xi, 0.f), (float)(W - 1));
            float yc = fminf(fmaxf(yi, 0.f), (float)(H - 1));
            acc  += wgt * valid * pl[(int)yc * W + (int)xc];
            mask += wgt * valid;
        }
    out[i] = (mask >= 1.f) ? acc : 0.f;
}

// 81-channel cost volume (md=4), mean over C, fused lrelu, writes at chOff
__global__ void costvol_kernel(const float* __restrict__ f1, const float* __restrict__ f2,
                               float* __restrict__ out, int C, int H, int W,
                               int CoutTot, int chOff, int n)
{
    int i = blockIdx.x * 256 + threadIdx.x;
    if (i >= n) return;
    int xx = i % W; int t = i / W;
    int yy = t % H; t /= H;
    int d = t % 81; int b = t / 81;
    int dy = d / 9 - 4, dx = d % 9 - 4;
    int sy = yy + dy, sx = xx + dx;
    float s = 0.f;
    if (sy >= 0 && sy < H && sx >= 0 && sx < W) {
        const float* a = f1 + ((size_t)b * C * H + yy) * W + xx;
        const float* bb = f2 + ((size_t)b * C * H + sy) * W + sx;
        size_t cs = (size_t)H * W;
#pragma unroll 4
        for (int c = 0; c < C; ++c)
            s += a[c * cs] * bb[c * cs];
    }
    s /= (float)C;
    s = LRELU(s);
    out[(((size_t)b * CoutTot + chOff + d) * H + yy) * W + xx] = s;
}

// ============================ host orchestration ===========================
struct ConvP { const float* w; const float* b; };

static inline int cdiv(int a, int b) { return (a + b - 1) / b; }

extern "C" void kernel_launch(void* const* d_in, const int* in_sizes, int n_in,
                              void* d_out, int out_size, void* d_ws, size_t ws_size,
                              hipStream_t stream)
{
    (void)in_sizes; (void)n_in; (void)out_size;
    const float* input1 = (const float*)d_in[0];
    const float* input2 = (const float*)d_in[1];

    // JAX tree-flatten order: params keys sorted 'ctx','est','feat'; 'b' < 'w'.
    ConvP ctxp[7], estp[5][6], featp[6][2];
    int idx = 2;
    for (int i = 0; i < 7; ++i) { ctxp[i].b = (const float*)d_in[idx++]; ctxp[i].w = (const float*)d_in[idx++]; }
    for (int l = 0; l < 5; ++l)
        for (int k = 0; k < 6; ++k) { estp[l][k].b = (const float*)d_in[idx++]; estp[l][k].w = (const float*)d_in[idx++]; }
    for (int f = 0; f < 6; ++f)
        for (int k = 0; k < 2; ++k) { featp[f][k].b = (const float*)d_in[idx++]; featp[f][k].w = (const float*)d_in[idx++]; }

    struct Arena {
        char* base; size_t cap, off;
        float* a(size_t elems) {
            size_t bts = (elems * sizeof(float) + 255) & ~(size_t)255;
            if (off + bts > cap) off = 0;   // last-resort wrap if ws undersized
            float* p = (float*)(base + off); off += bts; return p;
        }
    } ar{ (char*)d_ws, ws_size, 0 };

    const int B = 4;
    const int chsz[7] = { 3, 16, 32, 64, 96, 128, 196 };
    int Hs[7], Ws[7]; Hs[0] = 384; Ws[0] = 512;
    for (int f = 0; f < 6; ++f) { Hs[f + 1] = (Hs[f] - 1) / 2 + 1; Ws[f + 1] = (Ws[f] - 1) / 2 + 1; }

    auto conv = [&](const float* x, int Cin, int H, int W, ConvP p, int Cout,
                    float* out, int CoutTot, int chOff, int stride, int dil, bool lrelu) {
        int Hout = (H - 1) / stride + 1, Wout = (W - 1) / stride + 1;
        dim3 g(cdiv(B * Hout * Wout, 128), cdiv(Cout, 32));
        conv3x3_wmma<<<g, dim3(256), 0, stream>>>(x, p.w, p.b, out, B, Cin, H, W,
                                                  Cout, CoutTot, chOff, Hout, Wout,
                                                  stride, dil, (int)lrelu);
    };
    auto chancopy = [&](const float* src, float* dst, int C, int H, int W, int Cd, int off) {
        int n = B * C * H * W;
        chancopy_kernel<<<cdiv(n, 256), 256, 0, stream>>>(src, dst, C, H, W, Cd, off, n);
    };

    // ---- persistent buffers ----
    float* feat1[6]; float* feat2[6];
    for (int f = 0; f < 6; ++f) {
        size_t e = (size_t)B * chsz[f + 1] * Hs[f + 1] * Ws[f + 1];
        feat1[f] = ar.a(e); feat2[f] = ar.a(e);
    }
    float* flowL[5];
    for (int l = 0; l < 5; ++l)
        flowL[l] = ar.a((size_t)B * 2 * Hs[6 - l] * Ws[6 - l]);

    const size_t plane4 = (size_t)B * Hs[2] * Ws[2];   // level-4 pixel count (peak)
    float* slotA = ar.a((size_t)565 * plane4);         // dense-chain ping
    float* slotB = ar.a((size_t)565 * plane4);         // dense-chain pong
    float* slotC = ar.a((size_t)128 * plane4);         // ctx ping
    float* slotD = ar.a((size_t)128 * plane4);         // ctx pong
    float* x2wbuf = ar.a((size_t)32 * plane4);         // warped features (max at l=4)
    float* tmp = ar.a((size_t)B * 16 * Hs[1] * Ws[1]); // pyramid c1 scratch (max f=0)
    float* ctxo = ar.a((size_t)B * 2 * Hs[2] * Ws[2]); // context-net 2-ch output

    // ---- feature pyramids (both images) ----
    for (int img = 0; img < 2; ++img) {
        const float* cur = img ? input2 : input1;
        float** dst = img ? feat2 : feat1;
        int curC = 3;
        for (int f = 0; f < 6; ++f) {
            conv(cur, curC, Hs[f], Ws[f], featp[f][0], chsz[f + 1], tmp, chsz[f + 1], 0, 2, 1, true);
            conv(tmp, chsz[f + 1], Hs[f + 1], Ws[f + 1], featp[f][1], chsz[f + 1],
                 dst[f], chsz[f + 1], 0, 1, 1, true);
            cur = dst[f]; curC = chsz[f + 1];
        }
    }

    // ---- coarse-to-fine forward-flow levels ----
    const float wim = 512.f, him = 384.f;
    const int couts[5] = { 128, 128, 96, 64, 32 };
    for (int l = 0; l < 5; ++l) {
        int C = chsz[6 - l], H = Hs[6 - l], W = Ws[6 - l];
        const float* x1 = feat1[5 - l];
        const float* x2 = feat2[5 - l];
        float* flow = flowL[l];
        int n2 = B * 2 * H * W;
        const float* x2w;
        if (l == 0) {
            fill_kernel<<<cdiv(n2, 256), 256, 0, stream>>>(flow, 0.f, n2);
            x2w = x2;
        } else {
            int pH = Hs[6 - (l - 1)], pW = Ws[6 - (l - 1)];
            upsample_kernel<<<cdiv(n2, 256), 256, 0, stream>>>(flowL[l - 1], flow, 2, pH, pW, H, W, 1.f, n2);
            float cx = (float)(W - 1) / ((wim - 1.f) * 0.05f);
            float cy = (float)(H - 1) / ((him - 1.f) * 0.05f);
            int nw = B * C * H * W;
            warp_kernel<<<cdiv(nw, 256), 256, 0, stream>>>(x2, flow, x2wbuf, C, H, W, cx, cy, nw);
            x2w = x2wbuf;
        }
        // inp_f = [corr(81) | x1(C) | flow(2)]  (corr only at l==0)
        int Ci = (l == 0) ? 81 : (83 + C);
        int nc = B * 81 * H * W;
        costvol_kernel<<<cdiv(nc, 256), 256, 0, stream>>>(x1, x2w, slotA, C, H, W, Ci, 0, nc);
        if (l > 0) {
            chancopy(x1, slotA, C, H, W, Ci, 81);
            chancopy(flow, slotA, 2, H, W, Ci, 81 + C);
        }
        // dense flow-estimator chain
        float* cur = slotA; float* nxt = slotB; int curC = Ci;
        for (int k = 0; k < 5; ++k) {
            conv(cur, curC, H, W, estp[l][k], couts[k], nxt, curC + couts[k], 0, 1, 1, true);
            chancopy(cur, nxt, curC, H, W, curC + couts[k], couts[k]);
            float* t = cur; cur = nxt; nxt = t; curC += couts[k];
        }
        conv(cur, curC, H, W, estp[l][5], 2, flow, 2, 0, 1, 1, false);

        if (l == 4) {
            // context net on [xi_f | flow_f]
            float* cin = nxt;
            chancopy(cur, cin, curC, H, W, curC + 2, 0);
            chancopy(flow, cin, 2, H, W, curC + 2, curC);
            const int cch[7]  = { 128, 128, 128, 96, 64, 32, 2 };
            const int cdil[7] = { 1, 2, 4, 8, 16, 1, 1 };
            const float* a = cin; int ac = curC + 2;
            for (int i = 0; i < 7; ++i) {
                float* dstp = (i == 6) ? ctxo : ((i & 1) ? slotD : slotC);
                conv(a, ac, H, W, ctxp[i], cch[i], dstp, cch[i], 0, 1, cdil[i], i < 6);
                a = dstp; ac = cch[i];
            }
            add_kernel<<<cdiv(n2, 256), 256, 0, stream>>>(flow, ctxo, n2);
        }
    }

    // ---- final: upsample flow_f to full res, scale by 1/DIV_FLOW = 20 ----
    int no = B * 2 * 384 * 512;
    upsample_kernel<<<cdiv(no, 256), 256, 0, stream>>>(flowL[4], (float*)d_out,
                                                       2, Hs[2], Ws[2], 384, 512, 20.f, no);
}
